// CRF_BiLSTM_38740605010160
// MI455X (gfx1250) — compile-verified, matrix operand order checked by reading, not run on hardware
//
#include <hip/hip_runtime.h>
#include <hip/hip_bf16.h>

// ---------------------------------------------------------------------------
// CDNA5 BiLSTM + CRF.  H=1024, S=8192, T=48, START=46, END=47.
// ---------------------------------------------------------------------------
#define HD     1024
#define SEQ    8192
#define G4     4096          // 4*H
#define NTAG   48
#define TSTART 46
#define TEND   47
#define NEGV   -10000.0f
#define REC_NB 16            // blocks per direction in recurrence

typedef __attribute__((ext_vector_type(16))) __bf16 v16bf;
typedef __attribute__((ext_vector_type(8)))  float  v8f;

union FragCvt { uint4 u[2]; v16bf v; };

// float -> bf16 bits, round-to-nearest-even
__device__ __forceinline__ unsigned short f2bf(float f) {
    unsigned int u = __float_as_uint(f);
    unsigned int r = (u + 0x7fffu + ((u >> 16) & 1u)) >> 16;
    return (unsigned short)r;
}
__device__ __forceinline__ float sigm(float x) { return 1.0f / (1.0f + __expf(-x)); }

// Load a 16x32 bf16 A/B fragment per CDNA5 ISA layout:
//  lanes 0-15 : M=lane,    K = kb+{0..7} (v0-3), kb+{16..23} (v4-7)
//  lanes 16-31: M=lane-16, K = kb+{8..15},       kb+{24..31}
__device__ __forceinline__ v16bf load_frag(const unsigned short* base, int row_stride,
                                           int row, int kb, int lane) {
    int koff = (lane >> 4) * 8;
    const unsigned short* p = base + (size_t)row * row_stride + kb + koff;
    FragCvt c;
    c.u[0] = *reinterpret_cast<const uint4*>(p);
    c.u[1] = *reinterpret_cast<const uint4*>(p + 16);
    return c.v;
}

// ---------------------------------------------------------------------------
// Prep: fp32 -> bf16 conversions, bias combine, zero h0 / barriers.
// ---------------------------------------------------------------------------
__global__ __launch_bounds__(256) void prep_convert(
    const float* __restrict__ x,
    const float* __restrict__ wih_f, const float* __restrict__ wih_b,
    const float* __restrict__ bih_f, const float* __restrict__ bhh_f,
    const float* __restrict__ bih_b, const float* __restrict__ bhh_b,
    const float* __restrict__ wdense,
    unsigned short* __restrict__ xbf, unsigned short* __restrict__ wihbf,
    float* __restrict__ bcomb, unsigned short* __restrict__ wdbf,
    float* __restrict__ hbuf, unsigned int* __restrict__ bar) {
    size_t idx = (size_t)blockIdx.x * 256 + threadIdx.x;
    if (idx < (size_t)SEQ * HD)  xbf[idx] = f2bf(x[idx]);
    if (idx < (size_t)G4 * HD) {
        wihbf[idx]                    = f2bf(wih_f[idx]);
        wihbf[(size_t)G4 * HD + idx]  = f2bf(wih_b[idx]);
    }
    if (idx < G4) {
        bcomb[idx]      = bih_f[idx] + bhh_f[idx];
        bcomb[G4 + idx] = bih_b[idx] + bhh_b[idx];
    }
    if (idx < (size_t)NTAG * 2 * HD) wdbf[idx] = f2bf(wdense[idx]);
    if (idx < 2 * HD) hbuf[idx] = 0.0f;
    if (idx < 8)      bar[idx]  = 0u;
}

// Transpose W_hh to K-major, pack 2 bf16 per dword: wt2[dir][p][j] = {W[j][2p],W[j][2p+1]}
__global__ __launch_bounds__(256) void prep_whh(
    const float* __restrict__ whh_f, const float* __restrict__ whh_b,
    unsigned int* __restrict__ wt2) {
    size_t idx = (size_t)blockIdx.x * 256 + threadIdx.x;
    if (idx >= (size_t)2 * (HD / 2) * G4) return;
    int dir = (int)(idx / ((HD / 2) * G4));
    int rem = (int)(idx % ((HD / 2) * G4));
    int p = rem / G4, j = rem % G4;
    const float* W = dir ? whh_b : whh_f;
    unsigned short lo = f2bf(W[(size_t)j * HD + 2 * p]);
    unsigned short hi = f2bf(W[(size_t)j * HD + 2 * p + 1]);
    wt2[idx] = ((unsigned int)hi << 16) | lo;
}

// ---------------------------------------------------------------------------
// xi[dir][s][j] = bcomb[dir][j] + sum_k x[s or S-1-s][k] * W_ih[dir][j][k]
// Register-blocked WMMA: each wave owns a 32M x 64N block -> per K-step,
// 6 fragment loads feed 8 v_wmma (2 A frags x 4 B frags), K = 1024.
// ---------------------------------------------------------------------------
__global__ __launch_bounds__(256) void gemm_xi(
    const unsigned short* __restrict__ xbf, const unsigned short* __restrict__ wihbf,
    const float* __restrict__ bcomb, float* __restrict__ xi) {
    int wave = (blockIdx.x * 256 + (int)threadIdx.x) >> 5;
    int lane = threadIdx.x & 31;
    const int MB = SEQ / 32, NB = G4 / 64;           // 256, 64
    int dir = wave / (MB * NB);
    int rem = wave % (MB * NB);
    int mb = rem / NB, nb = rem % NB;
    const unsigned short* W = wihbf + (size_t)dir * G4 * HD;
    const float* bias = bcomb + (size_t)dir * G4;
    float* out = xi + (size_t)dir * SEQ * G4;

    int m0 = mb * 32, n0 = nb * 64;
    int arow0 = m0 + (lane & 15);
    int arow1 = m0 + 16 + (lane & 15);
    if (dir) { arow0 = SEQ - 1 - arow0; arow1 = SEQ - 1 - arow1; }  // reversed x
    int bcol = lane & 15;

    v8f acc[2][4];
#pragma unroll
    for (int ai = 0; ai < 2; ++ai)
#pragma unroll
        for (int bi = 0; bi < 4; ++bi) acc[ai][bi] = (v8f){};

    for (int kb = 0; kb < HD; kb += 32) {
        v16bf a0 = load_frag(xbf, HD, arow0, kb, lane);
        v16bf a1 = load_frag(xbf, HD, arow1, kb, lane);
        v16bf b0 = load_frag(W, HD, n0 +  0 + bcol, kb, lane);
        v16bf b1 = load_frag(W, HD, n0 + 16 + bcol, kb, lane);
        v16bf b2 = load_frag(W, HD, n0 + 32 + bcol, kb, lane);
        v16bf b3 = load_frag(W, HD, n0 + 48 + bcol, kb, lane);
        acc[0][0] = __builtin_amdgcn_wmma_f32_16x16x32_bf16(false, a0, false, b0, (short)0, acc[0][0], false, false);
        acc[0][1] = __builtin_amdgcn_wmma_f32_16x16x32_bf16(false, a0, false, b1, (short)0, acc[0][1], false, false);
        acc[0][2] = __builtin_amdgcn_wmma_f32_16x16x32_bf16(false, a0, false, b2, (short)0, acc[0][2], false, false);
        acc[0][3] = __builtin_amdgcn_wmma_f32_16x16x32_bf16(false, a0, false, b3, (short)0, acc[0][3], false, false);
        acc[1][0] = __builtin_amdgcn_wmma_f32_16x16x32_bf16(false, a1, false, b0, (short)0, acc[1][0], false, false);
        acc[1][1] = __builtin_amdgcn_wmma_f32_16x16x32_bf16(false, a1, false, b1, (short)0, acc[1][1], false, false);
        acc[1][2] = __builtin_amdgcn_wmma_f32_16x16x32_bf16(false, a1, false, b2, (short)0, acc[1][2], false, false);
        acc[1][3] = __builtin_amdgcn_wmma_f32_16x16x32_bf16(false, a1, false, b3, (short)0, acc[1][3], false, false);
    }
#pragma unroll
    for (int ai = 0; ai < 2; ++ai) {
        int mbase = m0 + ai * 16 + ((lane >> 4) * 8);
#pragma unroll
        for (int bi = 0; bi < 4; ++bi) {
            int n = n0 + bi * 16 + (lane & 15);
            float bv = bias[n];
#pragma unroll
            for (int v = 0; v < 8; ++v)
                out[(size_t)(mbase + v) * G4 + n] = acc[ai][bi][v] + bv;
        }
    }
}

// ---------------------------------------------------------------------------
// Persistent recurrence. grid = 2*REC_NB blocks x 256 threads.
// Block (dir, blk) owns h rows [blk*64, blk*64+64): computes their 4 gate rows.
// One atomic grid-sync per step among the direction's REC_NB blocks.
// ---------------------------------------------------------------------------
__global__ __launch_bounds__(256) void lstm_rec(
    const float* __restrict__ xi, const unsigned int* __restrict__ wt2,
    float* __restrict__ hbuf, unsigned short* __restrict__ hs,
    unsigned int* __restrict__ bar) {
    __shared__ float hlds[HD];
    __shared__ float gbuf[256];
    int dir = blockIdx.x / REC_NB;
    int blk = blockIdx.x % REC_NB;
    int t   = threadIdx.x;
    const float*        xid = xi  + (size_t)dir * SEQ * G4;
    const unsigned int* wt  = wt2 + (size_t)dir * (HD / 2) * G4;
    float* hd = hbuf + dir * HD;
    unsigned int* cnt   = bar + dir;
    unsigned int* phase = bar + 2 + dir;

    int chunk = t >> 6, li = t & 63;
    int grow  = chunk * HD + blk * 64 + li;          // gate row this thread computes
    float c = 0.0f;

    for (int step = 0; step < SEQ; ++step) {
        for (int i = t; i < HD; i += 256) hlds[i] = hd[i];
        __syncthreads();

        float acc = xid[(size_t)step * G4 + grow];
        const unsigned int* wp = wt + grow;
#pragma unroll 8
        for (int p = 0; p < HD / 2; ++p) {
            unsigned int w = wp[(size_t)p * G4];
            acc = fmaf(hlds[2 * p],     __uint_as_float(w << 16),          acc);
            acc = fmaf(hlds[2 * p + 1], __uint_as_float(w & 0xffff0000u),  acc);
        }
        gbuf[t] = acc;
        __syncthreads();

        if (t < 64) {
            float iv = sigm(gbuf[t]);
            float fv = sigm(gbuf[64 + t]);
            float gv = tanhf(gbuf[128 + t]);
            float ov = sigm(gbuf[192 + t]);
            c = fv * c + iv * gv;
            float h = ov * tanhf(c);
            int hidx = blk * 64 + t;
            hd[hidx] = h;
            int row = dir ? (SEQ - 1 - step) : step;  // backward outputs re-reversed
            hs[(size_t)row * (2 * HD) + dir * HD + hidx] = f2bf(h);
        }
        __syncthreads();
        __threadfence();
        if (t == 0) {
            unsigned int prev = atomicAdd(cnt, 1u);
            if (prev == REC_NB - 1) {
                *cnt = 0u;
                __threadfence();
                atomicAdd(phase, 1u);
            }
            while (atomicAdd(phase, 0u) < (unsigned int)(step + 1))
                __builtin_amdgcn_s_sleep(2);
        }
        __syncthreads();
    }
}

// ---------------------------------------------------------------------------
// feats[s][n] = b_dense[n] + sum_k hs[s][k] * W_dense[n][k], K = 2048, N = 48.
// Each wave handles one M-tile against all 3 N-tiles (A fragment reused 3x).
// ---------------------------------------------------------------------------
__global__ __launch_bounds__(256) void gemm_dense(
    const unsigned short* __restrict__ hsbf, const unsigned short* __restrict__ wdbf,
    const float* __restrict__ bd, float* __restrict__ feats) {
    int wave = (blockIdx.x * 256 + (int)threadIdx.x) >> 5;   // 512 waves
    int lane = threadIdx.x & 31;
    int arow = wave * 16 + (lane & 15);
    int bcol = lane & 15;
    v8f acc[3];
#pragma unroll
    for (int bi = 0; bi < 3; ++bi) acc[bi] = (v8f){};
    for (int kb = 0; kb < 2 * HD; kb += 32) {
        v16bf a  = load_frag(hsbf, 2 * HD, arow, kb, lane);
        v16bf b0 = load_frag(wdbf, 2 * HD,  0 + bcol, kb, lane);
        v16bf b1 = load_frag(wdbf, 2 * HD, 16 + bcol, kb, lane);
        v16bf b2 = load_frag(wdbf, 2 * HD, 32 + bcol, kb, lane);
        acc[0] = __builtin_amdgcn_wmma_f32_16x16x32_bf16(false, a, false, b0, (short)0, acc[0], false, false);
        acc[1] = __builtin_amdgcn_wmma_f32_16x16x32_bf16(false, a, false, b1, (short)0, acc[1], false, false);
        acc[2] = __builtin_amdgcn_wmma_f32_16x16x32_bf16(false, a, false, b2, (short)0, acc[2], false, false);
    }
    int mbase = wave * 16 + ((lane >> 4) * 8);
#pragma unroll
    for (int bi = 0; bi < 3; ++bi) {
        int n = bi * 16 + (lane & 15);
        float bv = bd[n];
#pragma unroll
        for (int v = 0; v < 8; ++v)
            feats[(size_t)(mbase + v) * NTAG + n] = acc[bi][v] + bv;
    }
}

// ---------------------------------------------------------------------------
// Viterbi: forward scan (transitions in LDS, bptr -> ws), then backtrack.
// out[0] = path_score, out[1+t] = (float)best_path[t].
// ---------------------------------------------------------------------------
__global__ __launch_bounds__(64) void viterbi(
    const float* __restrict__ feats, const float* __restrict__ trans,
    unsigned char* __restrict__ bptr, float* __restrict__ out, int out_size) {
    __shared__ float tl[NTAG * NTAG];
    __shared__ float fv[NTAG];
    __shared__ float nfv[NTAG];
    int t = threadIdx.x;
    for (int i = t; i < NTAG * NTAG; i += 64) tl[i] = trans[i];
    if (t < NTAG) fv[t] = (t == TSTART) ? 0.0f : NEGV;
    __syncthreads();
    for (int s = 0; s < SEQ; ++s) {
        if (t < NTAG) {
            float best = -3.4e38f; int bj = 0;
            for (int j = 0; j < NTAG; ++j) {
                float v = fv[j] + tl[t * NTAG + j];
                if (v > best) { best = v; bj = j; }
            }
            nfv[t] = best + feats[(size_t)s * NTAG + t];
            bptr[(size_t)s * NTAG + t] = (unsigned char)bj;
        }
        __syncthreads();
        if (t < NTAG) fv[t] = nfv[t];
        __syncthreads();
    }
    if (t == 0) {
        float best = -3.4e38f; int bl = 0;
        for (int i = 0; i < NTAG; ++i) {
            float v = fv[i] + tl[TEND * NTAG + i];
            if (v > best) { best = v; bl = i; }
        }
        if (out_size > 0) out[0] = best;
        int tag = bl;
        for (int s = SEQ - 1; s >= 0; --s) {
            if (1 + s < out_size) out[1 + s] = (float)tag;
            tag = (int)bptr[(size_t)s * NTAG + tag];
        }
    }
}

// ---------------------------------------------------------------------------
extern "C" void kernel_launch(void* const* d_in, const int* in_sizes, int n_in,
                              void* d_out, int out_size, void* d_ws, size_t ws_size,
                              hipStream_t stream) {
    const float* x      = (const float*)d_in[0];
    const float* wih_f  = (const float*)d_in[1];
    const float* whh_f  = (const float*)d_in[2];
    const float* bih_f  = (const float*)d_in[3];
    const float* bhh_f  = (const float*)d_in[4];
    const float* wih_b  = (const float*)d_in[5];
    const float* whh_b  = (const float*)d_in[6];
    const float* bih_b  = (const float*)d_in[7];
    const float* bhh_b  = (const float*)d_in[8];
    const float* wdense = (const float*)d_in[9];
    const float* bdense = (const float*)d_in[10];
    const float* trans  = (const float*)d_in[11];
    float* out = (float*)d_out;

    // workspace carve-up (256B aligned)
    char* ws = (char*)d_ws;
    size_t off = 0;
    auto carve = [&](size_t bytes) -> char* {
        char* p = ws + off;
        off = (off + bytes + 255) & ~(size_t)255;
        return p;
    };
    unsigned int*   bar   = (unsigned int*)  carve(256);
    float*          hbuf  = (float*)         carve((size_t)2 * HD * 4);
    unsigned short* xbf   = (unsigned short*)carve((size_t)SEQ * HD * 2);
    unsigned short* wihbf = (unsigned short*)carve((size_t)2 * G4 * HD * 2);
    float*          bcomb = (float*)         carve((size_t)2 * G4 * 4);
    unsigned short* wdbf  = (unsigned short*)carve((size_t)NTAG * 2 * HD * 2);
    unsigned int*   wt2   = (unsigned int*)  carve((size_t)2 * (HD / 2) * G4 * 4);
    float*          xi    = (float*)         carve((size_t)2 * SEQ * G4 * 4);
    unsigned short* hsbf  = (unsigned short*)carve((size_t)SEQ * 2 * HD * 2);
    float*          feats = (float*)         carve((size_t)SEQ * NTAG * 4);
    unsigned char*  bptr  = (unsigned char*) carve((size_t)SEQ * NTAG);
    (void)ws_size; (void)in_sizes; (void)n_in;

    // 1) conversions + zero init (largest domain: S*H elements)
    prep_convert<<<(SEQ * HD + 255) / 256, 256, 0, stream>>>(
        x, wih_f, wih_b, bih_f, bhh_f, bih_b, bhh_b, wdense,
        xbf, wihbf, bcomb, wdbf, hbuf, bar);
    // 2) W_hh transpose/pack
    prep_whh<<<(2 * (HD / 2) * G4 + 255) / 256, 256, 0, stream>>>(whh_f, whh_b, wt2);
    // 3) input projection GEMM (WMMA, 32x64 register-blocked): 2*256*64 waves
    gemm_xi<<<(2 * (SEQ / 32) * (G4 / 64)) / 8, 256, 0, stream>>>(xbf, wihbf, bcomb, xi);
    // 4) persistent bidirectional recurrence
    lstm_rec<<<2 * REC_NB, 256, 0, stream>>>(xi, wt2, hbuf, hsbf, bar);
    // 5) dense projection GEMM (WMMA, A reused across 3 N-tiles): 512 waves
    gemm_dense<<<(SEQ / 16) / 8, 256, 0, stream>>>(hsbf, wdbf, bdense, feats);
    // 6) Viterbi forward + backtrack
    viterbi<<<1, 64, 0, stream>>>(feats, trans, bptr, out, out_size);
}
